// EGNNLocal_36670430773270
// MI455X (gfx1250) — matrix operand order repaired and checked by reading, not compiled, not run on hardware
//
#include <hip/hip_runtime.h>
#include <math.h>

#define NPTS 8192
#define KNN  16
#define NE   (NPTS*KNN)   /* 131072 edges */
#define HD   256
#define NOBJ 8
#define EPSV 1e-8f

typedef __attribute__((ext_vector_type(16))) __bf16 v16bf;
typedef __attribute__((ext_vector_type(8)))  __bf16 v8bf;
typedef __attribute__((ext_vector_type(8)))  float  v8f;

union V16U { v16bf v; v8bf h[2]; };

__device__ __forceinline__ float dev_silu(float x) { return x / (1.f + __expf(-x)); }

// ---------------------------------------------------------------------------
// KNN: one thread per point, full scan, register-resident top-16 insertion.
// ---------------------------------------------------------------------------
__global__ void knn_kernel(const float* __restrict__ pc, const int* __restrict__ tag,
                           int* __restrict__ src) {
    int i = blockIdx.x * blockDim.x + threadIdx.x;
    if (i >= NPTS) return;
    float xi = pc[i*3], yi = pc[i*3+1], zi = pc[i*3+2];
    float sqi = xi*xi + yi*yi + zi*zi;
    int ti = tag[i];
    float bd[KNN]; int bi[KNN];
#pragma unroll
    for (int t = 0; t < KNN; ++t) { bd[t] = INFINITY; bi[t] = 0; }
    for (int j = 0; j < NPTS; ++j) {
        if (j == i) continue;
        if (tag[j] != ti) continue;
        float xj = pc[j*3], yj = pc[j*3+1], zj = pc[j*3+2];
        float sqj = xj*xj + yj*yj + zj*zj;
        float d = sqi + sqj - 2.f*(xi*xj + yi*yj + zi*zj);
        if (d < bd[KNN-1]) {
            float cd = d; int cj = j;
#pragma unroll
            for (int t = 0; t < KNN; ++t) {
                if (cd < bd[t]) {
                    float td = bd[t]; int tj = bi[t];
                    bd[t] = cd; bi[t] = cj; cd = td; cj = tj;
                }
            }
        }
    }
#pragma unroll
    for (int t = 0; t < KNN; ++t) src[i*KNN + t] = bi[t];
}

// ---------------------------------------------------------------------------
// h = emb_W * ||pc|| + emb_b ; x = pc
// ---------------------------------------------------------------------------
__global__ void init_h_kernel(const float* __restrict__ pc, const float* __restrict__ embW,
                              const float* __restrict__ embb, float* __restrict__ h,
                              __bf16* __restrict__ hb, float* __restrict__ x) {
    int idx = blockIdx.x * blockDim.x + threadIdx.x;
    if (idx >= NPTS*HD) return;
    int i = idx >> 8, c = idx & 255;
    float a = pc[i*3], b = pc[i*3+1], cc = pc[i*3+2];
    float nv = sqrtf(a*a + b*b + cc*cc);
    float v = embW[c]*nv + embb[c];
    h[idx] = v; hb[idx] = (__bf16)v;
    if (c < 3) x[i*3 + c] = pc[i*3 + c];
}

__global__ void edge_attr_kernel(const float* __restrict__ pc, const int* __restrict__ src,
                                 __bf16* __restrict__ eab) {
    int e = blockIdx.x * blockDim.x + threadIdx.x;
    if (e >= NE) return;
    int s = src[e], d = e >> 4;
    float dx = pc[s*3]-pc[d*3], dy = pc[s*3+1]-pc[d*3+1], dz = pc[s*3+2]-pc[d*3+2];
    eab[e] = (__bf16)sqrtf(dx*dx + dy*dy + dz*dz);
}

// Per-layer edge geometry: radial, normalized direction (+EPS, as reference).
__global__ void edge_geom_kernel(const float* __restrict__ x, const int* __restrict__ src,
                                 __bf16* __restrict__ radb, float* __restrict__ cdn) {
    int e = blockIdx.x * blockDim.x + threadIdx.x;
    if (e >= NE) return;
    int s = src[e], d = e >> 4;
    float dx = x[s*3]-x[d*3], dy = x[s*3+1]-x[d*3+1], dz = x[s*3+2]-x[d*3+2];
    float radial = dx*dx + dy*dy + dz*dz;
    radb[e] = (__bf16)radial;
    float inv = 1.f / sqrtf(radial);
    cdn[e*3+0] = dx*inv + EPSV;
    cdn[e*3+1] = dy*inv + EPSV;
    cdn[e*3+2] = dz*inv + EPSV;
}

// ---------------------------------------------------------------------------
// WMMA GEMM for e0, register-blocked 4Mx2N (64 rows x 32 cols per wave).
// A = [h[src] | h[dst] | radial | edge_attr | 0-pad] (K padded 514->544),
// gathered on the fly in three straight-line phases (no branches in loop).
// W bf16 [256 x 544] row-major (= B^T). silu epilogue -> bf16.
// ---------------------------------------------------------------------------
__global__ void gemm_e0_kernel(const __bf16* __restrict__ hb, const int* __restrict__ src,
                               const __bf16* __restrict__ radb, const __bf16* __restrict__ eab,
                               const __bf16* __restrict__ Wb, const float* __restrict__ bias,
                               __bf16* __restrict__ out) {
    const int lane = threadIdx.x & 31;
    const int wave = threadIdx.x >> 5;
    const int tid  = blockIdx.x * 8 + wave;   // over (NE/64)*(256/32)
    const int tn   = tid & 7;                 // 32-col group
    const int tm   = tid >> 3;                // 64-row group
    const int half = lane >> 4;
    const int l16  = lane & 15;
    const __bf16* hs[4];                      // h[src] rows (per-lane)
    const __bf16* hd[4];                      // h[dst] rows (wave-uniform per tile)
    __bf16 rad[4], ea[4];
#pragma unroll
    for (int i = 0; i < 4; ++i) {
        int e = tm*64 + i*16 + l16;
        hs[i] = hb + (size_t)src[e]*HD;
        hd[i] = hb + (size_t)(e >> 4)*HD;     // e>>4 == tm*4+i (uniform)
        rad[i] = radb[e]; ea[i] = eab[e];
    }
    const __bf16* wrow0 = Wb + (size_t)(tn*32 + l16) * 544;
    const __bf16* wrow1 = wrow0 + (size_t)16 * 544;
    v8f acc[4][2];
#pragma unroll
    for (int i = 0; i < 4; ++i)
#pragma unroll
        for (int j = 0; j < 2; ++j)
#pragma unroll
            for (int r = 0; r < 8; ++r) acc[i][j][r] = 0.f;

    // Phase 1: kb in [0,256) -- A chunks from h[src]
#pragma unroll
    for (int kb = 0; kb < 256; kb += 32) {
        v16bf b0 = *(const v16bf*)(wrow0 + kb + 16*half);
        v16bf b1 = *(const v16bf*)(wrow1 + kb + 16*half);
        V16U a[4];
#pragma unroll
        for (int i = 0; i < 4; ++i) {
            a[i].h[0] = *(const v8bf*)(hs[i] + kb + 16*half);
            a[i].h[1] = *(const v8bf*)(hs[i] + kb + 16*half + 8);
        }
#pragma unroll
        for (int i = 0; i < 4; ++i) {
            acc[i][0] = __builtin_amdgcn_wmma_f32_16x16x32_bf16(false, a[i].v, false, b0,
                                                                (short)0, acc[i][0], false, false);
            acc[i][1] = __builtin_amdgcn_wmma_f32_16x16x32_bf16(false, a[i].v, false, b1,
                                                                (short)0, acc[i][1], false, false);
        }
    }
    // Phase 2: kb in [256,512) -- A chunks from h[dst]
#pragma unroll
    for (int kb = 0; kb < 256; kb += 32) {
        v16bf b0 = *(const v16bf*)(wrow0 + 256 + kb + 16*half);
        v16bf b1 = *(const v16bf*)(wrow1 + 256 + kb + 16*half);
        V16U a[4];
#pragma unroll
        for (int i = 0; i < 4; ++i) {
            a[i].h[0] = *(const v8bf*)(hd[i] + kb + 16*half);
            a[i].h[1] = *(const v8bf*)(hd[i] + kb + 16*half + 8);
        }
#pragma unroll
        for (int i = 0; i < 4; ++i) {
            acc[i][0] = __builtin_amdgcn_wmma_f32_16x16x32_bf16(false, a[i].v, false, b0,
                                                                (short)0, acc[i][0], false, false);
            acc[i][1] = __builtin_amdgcn_wmma_f32_16x16x32_bf16(false, a[i].v, false, b1,
                                                                (short)0, acc[i][1], false, false);
        }
    }
    // Phase 3: kb = 512 tail -- [radial, edge_attr, 0...] only in half 0, slot 0/1
    {
        v16bf b0 = *(const v16bf*)(wrow0 + 512 + 16*half);
        v16bf b1 = *(const v16bf*)(wrow1 + 512 + 16*half);
#pragma unroll
        for (int i = 0; i < 4; ++i) {
            V16U a;
#pragma unroll
            for (int t = 0; t < 16; ++t) a.v[t] = (__bf16)0.f;
            if (half == 0) { a.v[0] = rad[i]; a.v[1] = ea[i]; }
            acc[i][0] = __builtin_amdgcn_wmma_f32_16x16x32_bf16(false, a.v, false, b0,
                                                                (short)0, acc[i][0], false, false);
            acc[i][1] = __builtin_amdgcn_wmma_f32_16x16x32_bf16(false, a.v, false, b1,
                                                                (short)0, acc[i][1], false, false);
        }
    }
#pragma unroll
    for (int i = 0; i < 4; ++i)
#pragma unroll
        for (int j = 0; j < 2; ++j) {
            int n = tn*32 + j*16 + l16;
            float bv = bias[n];
#pragma unroll
            for (int r = 0; r < 8; ++r) {
                int row = tm*64 + i*16 + r + 8*half;
                out[(size_t)row*HD + n] = (__bf16)dev_silu(acc[i][j][r] + bv);
            }
        }
}

// ---------------------------------------------------------------------------
// Generic WMMA GEMM, register-blocked 4Mx2N, K templated for full unroll:
// out = act(A[M,KD] @ W[Nd,KD]^T + bias).  M % 64 == 0, Nd % 32 == 0.
// ---------------------------------------------------------------------------
template<int KD>
__global__ void gemm_bf16_kernel(const __bf16* __restrict__ A, const __bf16* __restrict__ W,
                                 const float* __restrict__ bias, __bf16* __restrict__ outb,
                                 float* __restrict__ outf, int ntn2, int do_silu) {
    const int lane = threadIdx.x & 31;
    const int wave = threadIdx.x >> 5;
    const int tid  = blockIdx.x * 8 + wave;
    const int tn   = tid % ntn2;              // 32-col group
    const int tm   = tid / ntn2;              // 64-row group
    const int half = lane >> 4;
    const int l16  = lane & 15;
    const __bf16* arow  = A + (size_t)(tm*64 + l16) * KD;
    const __bf16* wrow0 = W + (size_t)(tn*32 + l16) * KD;
    const __bf16* wrow1 = wrow0 + (size_t)16 * KD;
    v8f acc[4][2];
#pragma unroll
    for (int i = 0; i < 4; ++i)
#pragma unroll
        for (int j = 0; j < 2; ++j)
#pragma unroll
            for (int r = 0; r < 8; ++r) acc[i][j][r] = 0.f;
#pragma unroll
    for (int kb = 0; kb < KD; kb += 32) {
        v16bf b0 = *(const v16bf*)(wrow0 + kb + 16*half);
        v16bf b1 = *(const v16bf*)(wrow1 + kb + 16*half);
        V16U a[4];
#pragma unroll
        for (int i = 0; i < 4; ++i) {
            const __bf16* ai = arow + (size_t)i*16*KD;
            a[i].h[0] = *(const v8bf*)(ai + kb + 16*half);
            a[i].h[1] = *(const v8bf*)(ai + kb + 16*half + 8);
        }
#pragma unroll
        for (int i = 0; i < 4; ++i) {
            acc[i][0] = __builtin_amdgcn_wmma_f32_16x16x32_bf16(false, a[i].v, false, b0,
                                                                (short)0, acc[i][0], false, false);
            acc[i][1] = __builtin_amdgcn_wmma_f32_16x16x32_bf16(false, a[i].v, false, b1,
                                                                (short)0, acc[i][1], false, false);
        }
    }
    const int Nd = ntn2*32;
#pragma unroll
    for (int i = 0; i < 4; ++i)
#pragma unroll
        for (int j = 0; j < 2; ++j) {
            int n = tn*32 + j*16 + l16;
            float bv = bias ? bias[n] : 0.f;
#pragma unroll
            for (int r = 0; r < 8; ++r) {
                int row = tm*64 + i*16 + r + 8*half;
                float v = acc[i][j][r] + bv;
                if (do_silu) v = dev_silu(v);
                if (outb) outb[(size_t)row*Nd + n] = (__bf16)v;
                if (outf) outf[(size_t)row*Nd + n] = v;
            }
        }
}

// segment_sum(ef, src) via atomics
__global__ void agg_kernel(const __bf16* __restrict__ ef, const int* __restrict__ src,
                           float* __restrict__ agg) {
    int idx = blockIdx.x * blockDim.x + threadIdx.x;
    if (idx >= NE*HD) return;
    int e = idx >> 8, c = idx & 255;
    atomicAdd(&agg[(size_t)src[e]*HD + c], (float)ef[idx]);
}

// c1 (H->1) dot + trans = cdn * val, segment-summed by src into dx
__global__ void c1_trans_kernel(const __bf16* __restrict__ t, const float* __restrict__ c1w,
                                const int* __restrict__ src, const float* __restrict__ cdn,
                                float* __restrict__ dx) {
    int e = blockIdx.x * blockDim.x + threadIdx.x;
    if (e >= NE) return;
    float sum = 0.f;
    const __bf16* tr = t + (size_t)e*HD;
#pragma unroll 4
    for (int k = 0; k < HD; k += 8) {
        v8bf c = *(const v8bf*)(tr + k);
#pragma unroll
        for (int j = 0; j < 8; ++j) sum += (float)c[j] * c1w[k+j];
    }
    int s = src[e];
    atomicAdd(&dx[s*3+0], cdn[e*3+0]*sum);
    atomicAdd(&dx[s*3+1], cdn[e*3+1]*sum);
    atomicAdd(&dx[s*3+2], cdn[e*3+2]*sum);
}

__global__ void x_update_kernel(float* __restrict__ x, const float* __restrict__ dx) {
    int i = blockIdx.x * blockDim.x + threadIdx.x;
    if (i < NPTS*3) x[i] += dx[i];
}

__global__ void ncat_kernel(const float* __restrict__ h, const float* __restrict__ agg,
                            __bf16* __restrict__ ncat) {
    int idx = blockIdx.x * blockDim.x + threadIdx.x;
    if (idx >= NPTS*512) return;
    int i = idx >> 9, c = idx & 511;
    float v = (c < HD) ? h[(size_t)i*HD + c] : agg[(size_t)i*HD + (c - HD)];
    ncat[idx] = (__bf16)v;
}

__global__ void h_update_kernel(float* __restrict__ h, __bf16* __restrict__ hb,
                                const float* __restrict__ n1o) {
    int idx = blockIdx.x * blockDim.x + threadIdx.x;
    if (idx >= NPTS*HD) return;
    float v = h[idx] + n1o[idx];
    h[idx] = v; hb[idx] = (__bf16)v;
}

__global__ void pool_s_kernel(const float* __restrict__ h, const int* __restrict__ tag,
                              float* __restrict__ s) {
    int idx = blockIdx.x * blockDim.x + threadIdx.x;
    if (idx >= NPTS*HD) return;
    int i = idx >> 8, c = idx & 255;
    atomicAdd(&s[(size_t)tag[i]*HD + c], h[idx]);
}

__global__ void pool_v_kernel(const float* __restrict__ x, const int* __restrict__ tag,
                              float* __restrict__ v) {
    int idx = blockIdx.x * blockDim.x + threadIdx.x;
    if (idx >= NPTS*3) return;
    int i = idx / 3, d = idx % 3;
    atomicAdd(&v[tag[i]*3 + d], x[idx]);
}

// Tiny mixer MLP: one block per object (288 threads)
__global__ void mix_kernel(const float* __restrict__ s, const float* __restrict__ v,
                           const float* __restrict__ m0W, const float* __restrict__ m0b,
                           const float* __restrict__ m1W, const float* __restrict__ m1b,
                           float* __restrict__ mix) {
    int obj = blockIdx.x;
    __shared__ float in[257];
    __shared__ float t[256];
    int tid = threadIdx.x;
    if (tid < 256) in[tid] = s[obj*HD + tid];
    if (tid == 256) {
        float a = v[obj*3], b = v[obj*3+1], c = v[obj*3+2];
        in[256] = sqrtf(a*a + b*b + c*c);
    }
    __syncthreads();
    if (tid < 256) {
        float acc = m0b[tid];
        const float* wr = m0W + tid*257;
        for (int k = 0; k < 257; ++k) acc += wr[k] * in[k];
        t[tid] = dev_silu(acc);
    }
    __syncthreads();
    if (tid < 272) {
        float acc = m1b[tid];
        const float* wr = m1W + tid*256;
        for (int k = 0; k < 256; ++k) acc += wr[k] * t[k];
        mix[obj*272 + tid] = tanhf(acc);
    }
}

__global__ void scodes_kernel(const float* __restrict__ s, const float* __restrict__ roW,
                              const float* __restrict__ rob, const float* __restrict__ mix,
                              float* __restrict__ out) {
    int idx = blockIdx.x * blockDim.x + threadIdx.x;
    if (idx >= NOBJ*HD) return;
    int obj = idx >> 8, c = idx & 255;
    float acc = rob[c];
    const float* wr = roW + c*HD;
    const float* sr = s + obj*HD;
    for (int k = 0; k < HD; ++k) acc += wr[k] * sr[k];
    out[idx] = acc * mix[obj*272 + c];
}

__global__ void vcodes_kernel(const float* __restrict__ v, const float* __restrict__ mix,
                              float* __restrict__ out) {
    int idx = blockIdx.x * blockDim.x + threadIdx.x;
    if (idx >= NOBJ*16*3) return;
    int obj = idx / 48, r = idx % 48, k = r / 3, d = r % 3;
    out[NOBJ*HD + idx] = v[obj*3 + d] * mix[obj*272 + 256 + k];
}

__global__ void zero_kernel(float* __restrict__ p, int n) {
    int i = blockIdx.x * blockDim.x + threadIdx.x;
    if (i < n) p[i] = 0.f;
}

__global__ void cvt_pad_kernel(const float* __restrict__ src, __bf16* __restrict__ dst,
                               int rows, int scols, int dcols) {
    int idx = blockIdx.x * blockDim.x + threadIdx.x;
    if (idx >= rows*dcols) return;
    int r = idx / dcols, c = idx % dcols;
    dst[idx] = (c < scols) ? (__bf16)src[(size_t)r*scols + c] : (__bf16)0.f;
}

// ---------------------------------------------------------------------------
extern "C" void kernel_launch(void* const* d_in, const int* in_sizes, int n_in,
                              void* d_out, int out_size, void* d_ws, size_t ws_size,
                              hipStream_t stream) {
    (void)in_sizes; (void)n_in; (void)out_size;
    const float* pc  = (const float*)d_in[0];
    const int*   tag = (const int*)d_in[1];
    const float* embW = (const float*)d_in[3];
    const float* embb = (const float*)d_in[4];
    const float *e0W[3], *e0b[3], *e1W[3], *e1b[3], *n0W[3], *n0b[3],
                *n1W[3], *n1b[3], *c0W[3], *c0b[3], *c1W[3];
    for (int l = 0; l < 3; ++l) {
        int b = 5 + l*11;
        e0W[l]=(const float*)d_in[b+0];  e0b[l]=(const float*)d_in[b+1];
        e1W[l]=(const float*)d_in[b+2];  e1b[l]=(const float*)d_in[b+3];
        n0W[l]=(const float*)d_in[b+4];  n0b[l]=(const float*)d_in[b+5];
        n1W[l]=(const float*)d_in[b+6];  n1b[l]=(const float*)d_in[b+7];
        c0W[l]=(const float*)d_in[b+8];  c0b[l]=(const float*)d_in[b+9];
        c1W[l]=(const float*)d_in[b+10];
    }
    const float* roW = (const float*)d_in[38];
    const float* rob = (const float*)d_in[39];
    const float* m0W = (const float*)d_in[40];
    const float* m0b = (const float*)d_in[41];
    const float* m1W = (const float*)d_in[42];
    const float* m1b = (const float*)d_in[43];

    size_t off = 0;
    auto alloc = [&](size_t bytes) -> void* {
        off = (off + 255) & ~(size_t)255;
        void* r = (char*)d_ws + off;
        off += bytes;
        return r;
    };
    int*    srcb = (int*)   alloc((size_t)NE*4);
    float*  h    = (float*) alloc((size_t)NPTS*HD*4);
    __bf16* hb   = (__bf16*)alloc((size_t)NPTS*HD*2);
    float*  x    = (float*) alloc((size_t)NPTS*3*4);
    float*  dx   = (float*) alloc((size_t)NPTS*3*4);
    __bf16* eab  = (__bf16*)alloc((size_t)NE*2);
    __bf16* radb = (__bf16*)alloc((size_t)NE*2);
    float*  cdn  = (float*) alloc((size_t)NE*3*4);
    __bf16* t0   = (__bf16*)alloc((size_t)NE*HD*2);
    __bf16* t1   = (__bf16*)alloc((size_t)NE*HD*2);
    float*  agg  = (float*) alloc((size_t)NPTS*HD*4);
    __bf16* ncat = (__bf16*)alloc((size_t)NPTS*512*2);
    __bf16* tn0  = (__bf16*)alloc((size_t)NPTS*HD*2);
    float*  n1o  = (float*) alloc((size_t)NPTS*HD*4);
    float*  sbuf = (float*) alloc((size_t)NOBJ*HD*4);
    float*  vbuf = (float*) alloc((size_t)NOBJ*3*4);
    float*  mixb = (float*) alloc((size_t)NOBJ*272*4);
    __bf16 *e0wb[3], *e1wb[3], *n0wb[3], *n1wb[3], *c0wb[3];
    for (int l = 0; l < 3; ++l) {
        e0wb[l] = (__bf16*)alloc((size_t)256*544*2);
        e1wb[l] = (__bf16*)alloc((size_t)256*256*2);
        n0wb[l] = (__bf16*)alloc((size_t)256*512*2);
        n1wb[l] = (__bf16*)alloc((size_t)256*256*2);
        c0wb[l] = (__bf16*)alloc((size_t)256*256*2);
    }
    if (off > ws_size) return;

    const int T = 256;
    auto blocks = [](long n, int t) { return (unsigned)((n + t - 1) / t); };

    // Weight conversion (bf16, e0 K-padded 514->544)
    for (int l = 0; l < 3; ++l) {
        cvt_pad_kernel<<<blocks(256*544, T), T, 0, stream>>>(e0W[l], e0wb[l], 256, 514, 544);
        cvt_pad_kernel<<<blocks(256*256, T), T, 0, stream>>>(e1W[l], e1wb[l], 256, 256, 256);
        cvt_pad_kernel<<<blocks(256*512, T), T, 0, stream>>>(n0W[l], n0wb[l], 256, 512, 512);
        cvt_pad_kernel<<<blocks(256*256, T), T, 0, stream>>>(n1W[l], n1wb[l], 256, 256, 256);
        cvt_pad_kernel<<<blocks(256*256, T), T, 0, stream>>>(c0W[l], c0wb[l], 256, 256, 256);
    }

    knn_kernel<<<blocks(NPTS, T), T, 0, stream>>>(pc, tag, srcb);
    init_h_kernel<<<blocks((long)NPTS*HD, T), T, 0, stream>>>(pc, embW, embb, h, hb, x);
    edge_attr_kernel<<<blocks(NE, T), T, 0, stream>>>(pc, srcb, eab);

    // 4Mx2N register blocking: wave tile = 64 rows x 32 cols, 8 waves/block
    const unsigned EDGE_GEMM_BLOCKS = (NE/64)*(HD/32)/8;    // 2048
    const unsigned NODE_GEMM_BLOCKS = (NPTS/64)*(HD/32)/8;  // 128

    for (int l = 0; l < 3; ++l) {
        zero_kernel<<<blocks(NPTS*3, T), T, 0, stream>>>(dx, NPTS*3);
        zero_kernel<<<blocks((long)NPTS*HD, T), T, 0, stream>>>(agg, NPTS*HD);
        edge_geom_kernel<<<blocks(NE, T), T, 0, stream>>>(x, srcb, radb, cdn);
        gemm_e0_kernel<<<EDGE_GEMM_BLOCKS, T, 0, stream>>>(hb, srcb, radb, eab,
                                                           e0wb[l], e0b[l], t0);
        gemm_bf16_kernel<256><<<EDGE_GEMM_BLOCKS, T, 0, stream>>>(t0, e1wb[l], e1b[l],
                                                                  t1, nullptr, 8, 1);
        agg_kernel<<<blocks((long)NE*HD, T), T, 0, stream>>>(t1, srcb, agg);
        gemm_bf16_kernel<256><<<EDGE_GEMM_BLOCKS, T, 0, stream>>>(t1, c0wb[l], c0b[l],
                                                                  t0, nullptr, 8, 1);
        c1_trans_kernel<<<blocks(NE, T), T, 0, stream>>>(t0, c1W[l], srcb, cdn, dx);
        x_update_kernel<<<blocks(NPTS*3, T), T, 0, stream>>>(x, dx);
        ncat_kernel<<<blocks((long)NPTS*512, T), T, 0, stream>>>(h, agg, ncat);
        gemm_bf16_kernel<512><<<NODE_GEMM_BLOCKS, T, 0, stream>>>(ncat, n0wb[l], n0b[l],
                                                                  tn0, nullptr, 8, 1);
        gemm_bf16_kernel<256><<<NODE_GEMM_BLOCKS, T, 0, stream>>>(tn0, n1wb[l], n1b[l],
                                                                  nullptr, n1o, 8, 0);
        h_update_kernel<<<blocks((long)NPTS*HD, T), T, 0, stream>>>(h, hb, n1o);
    }

    zero_kernel<<<blocks(NOBJ*HD, T), T, 0, stream>>>(sbuf, NOBJ*HD);
    zero_kernel<<<1, 32, 0, stream>>>(vbuf, NOBJ*3);
    pool_s_kernel<<<blocks((long)NPTS*HD, T), T, 0, stream>>>(h, tag, sbuf);
    pool_v_kernel<<<blocks(NPTS*3, T), T, 0, stream>>>(x, tag, vbuf);
    mix_kernel<<<NOBJ, 288, 0, stream>>>(sbuf, vbuf, m0W, m0b, m1W, m1b, mixb);
    scodes_kernel<<<blocks(NOBJ*HD, T), T, 0, stream>>>(sbuf, roW, rob, mixb, (float*)d_out);
    vcodes_kernel<<<blocks(NOBJ*16*3, T), T, 0, stream>>>(vbuf, mixb, (float*)d_out);
}
